// DGM_d_77421080477832
// MI455X (gfx1250) — compile-verified, matrix operand order checked by reading, not run on hardware
//
#include <hip/hip_runtime.h>
#include <cstdint>

// ---------------------------------------------------------------------------
// DGM fused kernel for MI455X (gfx1250): embed GEMM (WMMA f16->f32),
// fused Gram-distance + Gumbel top-K with TDM noise streaming.
// ---------------------------------------------------------------------------

typedef __attribute__((ext_vector_type(16))) _Float16     v16h;
typedef __attribute__((ext_vector_type(8)))  float        v8f;
typedef __attribute__((ext_vector_type(4)))  unsigned int u32x4;
typedef __attribute__((ext_vector_type(8)))  int          i32x8;
typedef __attribute__((ext_vector_type(4)))  int          i32x4;

#define USE_TDM 1

static constexpr int   B_   = 4;
static constexpr int   N_   = 4096;
static constexpr int   D_   = 64;
static constexpr int   K_   = 16;
static constexpr float CLAMP_X  = 1.0e6f;
static constexpr float CLAMP_LO = -1.0e10f;
static constexpr float CLAMP_HI = 1.0e10f;

// out layout: xe [b*n*d] | edges0 [b*n*K] | edges1 [b*n*K] | logprobs [b*n*K]
static constexpr size_t XE_CNT  = (size_t)B_ * N_ * D_;       // 1048576
static constexpr size_t E_CNT   = (size_t)B_ * N_ * K_;       // 262144

// A-fragment K mapping for v_wmma_f32_16x16x32_f16 (and mirrored for B):
// lanes 0-15 hold K={0..7,16..23}, lanes 16-31 hold K={8..15,24..31}.
__device__ __forceinline__ int kmap(int e, int lane) {
    return (e & 7) + ((e >> 3) << 4) + ((lane >> 4) << 3);
}

// ---------------------------------------------------------------------------
// Kernel 1: xe = clip( clip(x) @ W ).  One wave per 16x16 output tile.
// ---------------------------------------------------------------------------
__global__ __launch_bounds__(128) void embed_kernel(const float* __restrict__ x,
                                                    const float* __restrict__ W,
                                                    float* __restrict__ xe) {
    const int lane = threadIdx.x & 31;
    const int ct   = threadIdx.x >> 5;     // column tile 0..3 (of 64 cols)
    const int rt   = blockIdx.x;           // row tile 0..1023 (of 16384 rows)
    const int m    = lane & 15;
    const int khi  = (lane >> 4) << 3;

    v16h a0, a1, b0, b1;
#pragma unroll
    for (int e = 0; e < 16; ++e) {
        const int k  = kmap(e, lane);
        float xa0 = x[(size_t)(rt * 16 + m) * D_ + k];
        float xa1 = x[(size_t)(rt * 16 + m) * D_ + k + 32];
        xa0 = fminf(fmaxf(xa0, -CLAMP_X), CLAMP_X);
        xa1 = fminf(fmaxf(xa1, -CLAMP_X), CLAMP_X);
        a0[e] = (_Float16)xa0;
        a1[e] = (_Float16)xa1;
        const int c = ct * 16 + m;         // B column = lane&15
        b0[e] = (_Float16)W[(size_t)k * D_ + c];
        b1[e] = (_Float16)W[(size_t)(k + 32) * D_ + c];
    }
    v8f acc = {};
    acc = __builtin_amdgcn_wmma_f32_16x16x32_f16(false, a0, false, b0,
                                                 (short)0, acc, false, false);
    acc = __builtin_amdgcn_wmma_f32_16x16x32_f16(false, a1, false, b1,
                                                 (short)0, acc, false, false);
#pragma unroll
    for (int j = 0; j < 8; ++j) {
        const int mm = j + khi;
        float v = fminf(fmaxf(acc[j], -CLAMP_X), CLAMP_X);
        xe[(size_t)(rt * 16 + mm) * D_ + ct * 16 + (lane & 15)] = v;
    }
}

// ---------------------------------------------------------------------------
// Kernel 2: sq[r] = sum_k xe[r][k]^2   (16384 rows; L2-resident traffic)
// ---------------------------------------------------------------------------
__global__ __launch_bounds__(256) void sq_kernel(const float* __restrict__ xe,
                                                 float* __restrict__ sq) {
    const int r = blockIdx.x * 256 + threadIdx.x;
    float s = 0.0f;
#pragma unroll
    for (int k = 0; k < D_; ++k) {
        float v = xe[(size_t)r * D_ + k];
        s += v * v;
    }
    sq[r] = s;
}

// ---------------------------------------------------------------------------
// Kernel 3: fused Gram distance + Gumbel perturb + streaming top-16 per row.
// Grid: b * (n/16) workgroups. Block: 512 threads (16 waves).
// ---------------------------------------------------------------------------
__global__ __launch_bounds__(512) void dgm_topk_kernel(const float* __restrict__ xeG,
                                                       const float* __restrict__ sqG,
                                                       const float* __restrict__ tempP,
                                                       const float* __restrict__ noise,
                                                       float* __restrict__ out) {
    __shared__ float sXe[16 * 64];          // row tile of xe (f32)
    __shared__ float sqrow[16];             // row squared norms
    __shared__ float candVal[16][288];      // 256 candidates + 16 incumbents
    __shared__ int   candIdx[16][288];
    __shared__ float sNoise[16 * 256];      // TDM-staged noise tile

    const int tid  = threadIdx.x;
    const int lane = tid & 31;
    const int wv   = tid >> 5;              // wave id 0..15 == owned row
    const int bIdx = blockIdx.x >> 8;       // batch
    const int r0   = (blockIdx.x & 255) * 16;
    const int khi  = (lane >> 4) << 3;

    // ---- Phase 0: stage row tile, norms, incumbents --------------------
    for (int i = tid; i < 16 * 64; i += 512)
        sXe[i] = xeG[((size_t)bIdx * N_ + r0) * D_ + i];
    __syncthreads();

    if (tid < 16) {
        float s = 0.0f;
#pragma unroll
        for (int k = 0; k < D_; ++k) { float v = sXe[tid * 64 + k]; s += v * v; }
        sqrow[tid] = s;
    }
    if (tid < 256) {
        candVal[tid >> 4][256 + (tid & 15)] = __builtin_inff();
        candIdx[tid >> 4][256 + (tid & 15)] = -1;
    }

    // A fragments (same 16 rows for every wave)
    v16h a0, a1;
#pragma unroll
    for (int e = 0; e < 16; ++e) {
        const int k = kmap(e, lane);
        a0[e] = (_Float16)sXe[(lane & 15) * 64 + k];
        a1[e] = (_Float16)sXe[(lane & 15) * 64 + k + 32];
    }

    float tc   = fminf(fmaxf(tempP[0], -5.0f), 5.0f);
    const float expT = __expf(tc);
    __syncthreads();

    float selVal = __builtin_inff();
    int   selIdx = 0;

    // ---- Column-block loop: 16 blocks of 256 columns -------------------
    for (int cb = 0; cb < 16; ++cb) {
#if USE_TDM && __has_builtin(__builtin_amdgcn_tensor_load_to_lds)
        if (wv == 0) {
            // Tensor DMA: 16 x 256 f32 tile of noise -> sNoise (LDS)
            const unsigned ldsOff = (unsigned)(uintptr_t)(&sNoise[0]);
            const unsigned long long ga = (unsigned long long)(uintptr_t)
                (noise + (((size_t)bIdx * N_ + r0) * N_ + (size_t)cb * 256));
            u32x4 g0;
            g0[0] = 1u;                                   // count=1, user mode
            g0[1] = ldsOff;                               // lds_addr
            g0[2] = (unsigned)(ga & 0xFFFFFFFFull);       // global_addr lo
            g0[3] = (unsigned)((ga >> 32) & 0x1FFFFFFull) // global_addr hi
                    | (2u << 30);                         // type = image
            i32x8 g1;
            g1[0] = (int)(2u << 16);                      // data_size = 4B
            g1[1] = (int)((unsigned)(N_ & 0xFFFF) << 16); // tensor_dim0 lo16
            g1[2] = (int)((unsigned)(N_ & 0xFFFF) << 16); // dim0 hi16=0 | dim1 lo16
            g1[3] = (int)(256u << 16);                    // dim1 hi16=0 | tile_dim0=256
            g1[4] = 16;                                   // tile_dim1=16, tile_dim2=0
            g1[5] = N_;                                   // tensor_dim0_stride lo32
            g1[6] = 0;
            g1[7] = 0;
            i32x4 gz4 = {0, 0, 0, 0};
            i32x8 gz8 = {0, 0, 0, 0, 0, 0, 0, 0};
            __builtin_amdgcn_tensor_load_to_lds(g0, g1, gz4, gz4, gz8, 0);
        }
#endif
        // B fragments: 16 columns of xe for this wave (L2-resident gathers)
        const int c0 = cb * 256 + wv * 16;                // global column base
        v16h b0, b1;
#pragma unroll
        for (int e = 0; e < 16; ++e) {
            const int k = kmap(e, lane);
            const size_t crow = (size_t)bIdx * N_ + c0 + (lane & 15);
            b0[e] = (_Float16)xeG[crow * D_ + k];
            b1[e] = (_Float16)xeG[crow * D_ + k + 32];
        }
        v8f acc = {};
        acc = __builtin_amdgcn_wmma_f32_16x16x32_f16(false, a0, false, b0,
                                                     (short)0, acc, false, false);
        acc = __builtin_amdgcn_wmma_f32_16x16x32_f16(false, a1, false, b1,
                                                     (short)0, acc, false, false);

#if USE_TDM && __has_builtin(__builtin_amdgcn_tensor_load_to_lds)
        if (wv == 0) __builtin_amdgcn_s_wait_tensorcnt(0);
#else
        for (int i = tid; i < 16 * 256; i += 512)
            sNoise[i] = noise[(((size_t)bIdx * N_ + r0 + (i >> 8)) * N_)
                              + (size_t)cb * 256 + (i & 255)];
#endif
        __syncthreads();   // TDM data visible; previous selection complete

        // ---- lq = clip(D)*expT - log(-log(noise)); deposit candidates --
        const float sqcL = sqG[(size_t)bIdx * N_ + c0 + (lane & 15)];
        const int   cw   = wv * 16 + (lane & 15);         // slot in cand row
#pragma unroll
        for (int j = 0; j < 8; ++j) {
            const int mm = j + khi;
            float d = sqrow[mm] + sqcL - 2.0f * acc[j];
            d = fmaxf(d, 0.0f);
            if (__builtin_isnan(d) || __builtin_isinf(d)) d = 1.0e6f;
            float logits = fminf(fmaxf(d, CLAMP_LO), CLAMP_HI) * expT;
            float u  = sNoise[mm * 256 + cw];
            float lq = logits - __logf(-__logf(u));
            if (__builtin_isnan(lq) || __builtin_isinf(lq)) lq = CLAMP_LO;
            candVal[mm][cw] = lq;
            candIdx[mm][cw] = c0 + (lane & 15);
        }
        __syncthreads();

        // ---- wave wv selects top-16 smallest of row wv (272 entries) ---
        float rv[9];
        unsigned consumed = 0;
#pragma unroll
        for (int i = 0; i < 9; ++i) {
            const int p = lane + 32 * i;
            rv[i] = (p < 272) ? candVal[wv][p] : __builtin_inff();
        }
#pragma unroll
        for (int k = 0; k < K_; ++k) {
            float bv = __builtin_inff();
            int   bp = -1;
#pragma unroll
            for (int i = 0; i < 9; ++i) {
                const int p = lane + 32 * i;
                if (p < 272 && !((consumed >> i) & 1u)) {
                    if (rv[i] < bv) { bv = rv[i]; bp = p; }
                }
            }
            for (int off = 16; off > 0; off >>= 1) {      // wave32 argmin
                float ov = __shfl_xor(bv, off, 32);
                int   op = __shfl_xor(bp, off, 32);
                if (ov < bv || (ov == bv && (unsigned)op < (unsigned)bp)) {
                    bv = ov; bp = op;
                }
            }
            const int gIdx = candIdx[wv][bp];             // broadcast read
            if ((bp & 31) == lane) consumed |= 1u << (bp >> 5);
            if (lane == k) { selVal = bv; selIdx = gIdx; }
        }
        // write incumbents back for next block (lane k holds k-th best)
        if (lane < 16) {
            candVal[wv][256 + lane] = selVal;
            candIdx[wv][256 + lane] = selIdx;
        }
    }

    // ---- Final outputs: edges + logprobs for row r0+wv ------------------
    const int gr = bIdx * N_ + r0 + wv;                   // global node id
    if (lane < K_) {
        const size_t e = (size_t)gr * K_ + lane;
        out[XE_CNT + e]             = (float)gr;                      // edges[0]
        out[XE_CNT + E_CNT + e]     = (float)(bIdx * N_ + selIdx);    // edges[1]
        out[XE_CNT + 2 * E_CNT + e] = fminf(fmaxf(selVal, CLAMP_LO), 0.0f);
    }
}

// ---------------------------------------------------------------------------
extern "C" void kernel_launch(void* const* d_in, const int* in_sizes, int n_in,
                              void* d_out, int out_size, void* d_ws, size_t ws_size,
                              hipStream_t stream) {
    (void)in_sizes; (void)n_in; (void)out_size; (void)ws_size;
    const float* x     = (const float*)d_in[0];
    const float* W     = (const float*)d_in[1];
    const float* temp  = (const float*)d_in[2];
    const float* noise = (const float*)d_in[3];
    float* out = (float*)d_out;
    float* sqW = (float*)d_ws;                 // 16384 floats of scratch

    // 1) xe = clip(clip(x) @ W): 1024 row-tiles x 4 col-tile waves
    embed_kernel<<<dim3((B_ * N_) / 16), dim3(128), 0, stream>>>(x, W, out);
    // 2) row squared norms
    sq_kernel<<<dim3((B_ * N_) / 256), dim3(256), 0, stream>>>(out, sqW);
    // 3) fused distance + Gumbel top-K: one WG per 16-row tile
    dgm_topk_kernel<<<dim3(B_ * (N_ / 16)), dim3(512), 0, stream>>>(
        out, sqW, temp, noise, out);
}
